// MultiHeadAttention_85469849190480
// MI455X (gfx1250) — compile-verified, matrix operand order checked by reading, not run on hardware
//
#include <hip/hip_runtime.h>
#include <hip/hip_bf16.h>
#include <cstdint>

// ---------------------------------------------------------------------------
// MultiHeadAttention for MI455X (gfx1250), wave32 + WMMA bf16 path.
// B=8, S=1024, D_MODEL=512, H=8, DEPTH=64.
// d_out = [ out (B*S*512 f32) | attn (B*H*S*S f32) ]
//
// Pipeline:
//   convert_bf16 : q,k,v f32 -> bf16 (B,S,D) (one-shot streaming)
//   prep_weights : Wq/Wk/Wv/Wo f32 (k,n) -> bf16 (n,k) transposed (one-shot)
//   proj x3      : qh,kh (B,H,S,DP) bf16 ; vh transposed (B,H,DP,S) bf16
//                  (B tile: double-buffered global_load_async_to_lds_b128)
//   attn         : WMMA logits -> dual softmax (f32) * adj -> attn (d_out)
//                  -> WMMA attn@V -> ctx bf16
//   outproj      : ctx @ Wo + bo -> out f32 (d_out)
// All GEMM fragments are pure 16B loads (wave32 bf16 fragment layout's two
// 4-dword K-groups are contiguous in a K-major row).
// ---------------------------------------------------------------------------

#define B_   8
#define S_   1024
#define DM_  512
#define H_   8
#define DP_  64

typedef __bf16 bf16_t;
typedef __attribute__((ext_vector_type(16))) __bf16    v16bf;
typedef __attribute__((ext_vector_type(8)))  float     v8f;
typedef __attribute__((ext_vector_type(4)))  float     f32x4;
typedef __attribute__((ext_vector_type(4)))  unsigned  u32x4;
typedef __attribute__((ext_vector_type(2)))  unsigned  u32x2;

union BF16x16 { v16bf v; unsigned u[8]; u32x4 q[2]; };

// 16-bit A/B fragment layout (ISA 7.12.2, wave32):
// lane half = lane>>4 ; pair slots 0..3 hold K window [8*half, 8*half+8),
// pair slots 4..7 hold [16+8*half, 16+8*half+8). Two b128 loads per fragment.
static __device__ __forceinline__ v16bf frag_from_bf16(const bf16_t* __restrict__ row,
                                                       int half) {
  const u32x4* pv = (const u32x4*)row;     // 16B-aligned by construction
  BF16x16 r;
  r.q[0] = pv[half];        // K = 8*half .. 8*half+7
  r.q[1] = pv[2 + half];    // K = 16+8*half .. 16+8*half+7
  return r.v;
}

static __device__ __forceinline__ v8f wmma_bf16f32(v16bf a, v16bf b, v8f c) {
  return __builtin_amdgcn_wmma_f32_16x16x32_bf16(false, a, false, b, (short)0, c,
                                                 false, false);
}

// Async global->LDS 16B copy (CDNA5 ASYNCcnt path), one lane = one b128.
static __device__ __forceinline__ void async_copy_b128(unsigned lds_addr,
                                                       const bf16_t* gptr) {
  asm volatile("global_load_async_to_lds_b128 %0, %1, off"
               :: "v"(lds_addr), "v"((unsigned long long)(uintptr_t)gptr)
               : "memory");
}
static __device__ __forceinline__ void wait_async_le0() {
  asm volatile("s_wait_asynccnt 0x0" ::: "memory");
}
static __device__ __forceinline__ void wait_async_le1() {
  asm volatile("s_wait_asynccnt 0x1" ::: "memory");
}

// ---------------------------------------------------------------------------
// Kernel A: stream-convert f32 -> bf16 (q, k, v selected by blockIdx.z).
// One float4 load + one b64 store per thread.
// ---------------------------------------------------------------------------
__global__ void convert_bf16(const float* __restrict__ X0, const float* __restrict__ X1,
                             const float* __restrict__ X2,
                             bf16_t* __restrict__ Y0, bf16_t* __restrict__ Y1,
                             bf16_t* __restrict__ Y2) {
  const float* X; bf16_t* Y;
  switch (blockIdx.z) {
    case 0:  X = X0; Y = Y0; break;
    case 1:  X = X1; Y = Y1; break;
    default: X = X2; Y = Y2; break;
  }
  size_t i = ((size_t)blockIdx.x * 256 + threadIdx.x) * 4;
  f32x4 f = *(const f32x4*)(X + i);
  union { bf16_t b[4]; u32x2 u; } o;
  o.b[0] = (bf16_t)f.x; o.b[1] = (bf16_t)f.y;
  o.b[2] = (bf16_t)f.z; o.b[3] = (bf16_t)f.w;
  *(u32x2*)(Y + i) = o.u;
}

// ---------------------------------------------------------------------------
// Kernel B: transpose+convert 512x512 f32 weight (k,n) -> bf16 (n,k).
// ---------------------------------------------------------------------------
__global__ void prep_weights(const float* __restrict__ W0, const float* __restrict__ W1,
                             const float* __restrict__ W2, const float* __restrict__ W3,
                             bf16_t* __restrict__ T0, bf16_t* __restrict__ T1,
                             bf16_t* __restrict__ T2, bf16_t* __restrict__ T3) {
  const float* W; bf16_t* T;
  switch (blockIdx.z) {
    case 0:  W = W0; T = T0; break;
    case 1:  W = W1; T = T1; break;
    case 2:  W = W2; T = T2; break;
    default: W = W3; T = T3; break;
  }
  __shared__ bf16_t tile[64][65];  // +1 pad against bank conflicts
  const int n0 = blockIdx.x * 64, k0 = blockIdx.y * 64;
  const int t = threadIdx.x;
#pragma unroll
  for (int i = 0; i < 16; ++i) {
    int idx = t + 256 * i;                 // 0..4095
    int kk = idx >> 6, nn = idx & 63;      // coalesced along n
    tile[kk][nn] = (bf16_t)W[(size_t)(k0 + kk) * DM_ + n0 + nn];
  }
  __syncthreads();
#pragma unroll
  for (int i = 0; i < 16; ++i) {
    int idx = t + 256 * i;
    int nn = idx >> 6, kk = idx & 63;      // coalesced along k
    T[(size_t)(n0 + nn) * DM_ + k0 + kk] = tile[kk][nn];
  }
}

// ---------------------------------------------------------------------------
// Kernel 1: Y = Xb(8192x512 bf16) @ W + bias, bf16 out, W pre-transposed bf16.
// Block 256 thr / 8 waves; block tile M=128,N=64; wave tile 16x64.
// B tile (4KB) staged by 256 async b128 copies, double-buffered on ASYNCcnt.
// ---------------------------------------------------------------------------
__global__ void proj_kernel(const bf16_t* __restrict__ Xb, const bf16_t* __restrict__ WT,
                            const float* __restrict__ bias, bf16_t* __restrict__ Y,
                            int transpose_v) {
  __shared__ bf16_t ldsB[2][64 * 32];   // [buf][n][k], K-contiguous 64B rows
  const int n0 = blockIdx.x * 64;
  const int m0 = blockIdx.y * 128;
  const int t = threadIdx.x;
  const int wave = t >> 5, lane = t & 31;
  const int half = lane >> 4, l16 = lane & 15;

  const int rown = t >> 2, kc = (t & 3) * 8;   // thread's 16B chunk of tile
  const unsigned chunk = (unsigned)(rown * 64 + kc * 2);
  const unsigned ldsAddr0 = (unsigned)(uintptr_t)(&ldsB[0][0]) + chunk;
  const unsigned ldsAddr1 = (unsigned)(uintptr_t)(&ldsB[1][0]) + chunk;
  const bf16_t* wrow = WT + (size_t)(n0 + rown) * DM_ + kc;

  async_copy_b128(ldsAddr0, wrow);             // prologue: tile k0=0 -> buf0
  int cur = 0;
  v8f acc[4] = {};
  for (int k0 = 0; k0 < DM_; k0 += 32, cur ^= 1) {
    if (k0 + 32 < DM_) {
      async_copy_b128(cur ? ldsAddr0 : ldsAddr1, wrow + k0 + 32);
      wait_async_le1();                        // retire copy of buf[cur] only
    } else {
      wait_async_le0();
    }
    __syncthreads();                           // all waves' buf[cur] visible

    const bf16_t* arow = Xb + (size_t)(m0 + wave * 16 + l16) * DM_ + k0;
    v16bf a = frag_from_bf16(arow, half);
#pragma unroll
    for (int nt = 0; nt < 4; ++nt) {
      v16bf b = frag_from_bf16(&ldsB[cur][(nt * 16 + l16) * 32], half);
      acc[nt] = wmma_bf16f32(a, b, acc[nt]);
    }
    __syncthreads();                           // done reading buf[cur]
  }

  // C layout: lane l -> col = l&15, rows = 8*(l>>4)+r.
#pragma unroll
  for (int nt = 0; nt < 4; ++nt) {
    int n = n0 + nt * 16 + l16;
    int h = n >> 6, dp = n & 63;
    float bia = bias[n];
#pragma unroll
    for (int r = 0; r < 8; ++r) {
      int m = m0 + wave * 16 + 8 * half + r;
      int bb = m >> 10, s = m & (S_ - 1);
      float val = acc[nt][r] + bia;
      size_t dst = transpose_v
          ? (((size_t)(bb * H_ + h) * DP_ + dp) * S_ + s)
          : (((size_t)(bb * H_ + h) * S_ + s) * DP_ + dp);
      Y[dst] = (bf16_t)val;
    }
  }
}

// ---------------------------------------------------------------------------
// Kernel 2: attention for one (b, h, 16-query-row block).
// ---------------------------------------------------------------------------
__global__ void attn_kernel(const bf16_t* __restrict__ qh, const bf16_t* __restrict__ kh,
                            const bf16_t* __restrict__ vhT,
                            const float* __restrict__ mask, const float* __restrict__ adj,
                            const float* __restrict__ dist,
                            float* __restrict__ attn_out, bf16_t* __restrict__ ctx) {
  __shared__ float scores[16 * S_];          // 64 KB f32 logits
  __shared__ float ctxpart[8 * 16 * DP_];    // 32 KB; aliased as bf16 pAttn[16*S_]
  __shared__ float red[2][16][16];
  bf16_t* pAttn = (bf16_t*)ctxpart;

  const int q0 = blockIdx.x * 16;
  const int h = blockIdx.y, b = blockIdx.z;
  const int bh = b * H_ + h;
  const int t = threadIdx.x;
  const int wave = t >> 5, lane = t & 31;
  const int half = lane >> 4, l16 = lane & 15;

  // ---- Phase 1: logits (each wave covers 8 of the 64 key tiles) ----
  const bf16_t* qrow = qh + ((size_t)bh * S_ + q0 + l16) * DP_;
  v16bf aq0 = frag_from_bf16(qrow, half);        // depth 0..31
  v16bf aq1 = frag_from_bf16(qrow + 32, half);   // depth 32..63

  for (int nt = wave; nt < S_ / 16; nt += 8) {
    int n0 = nt * 16;
    const bf16_t* krow = kh + ((size_t)bh * S_ + n0 + l16) * DP_;
    v16bf b0 = frag_from_bf16(krow, half);
    v16bf b1 = frag_from_bf16(krow + 32, half);
    v8f c = {};
    c = wmma_bf16f32(aq0, b0, c);
    c = wmma_bf16f32(aq1, b1, c);
    int col = n0 + l16;
    float nm = mask[b * S_ + col] * -1e9f;
#pragma unroll
    for (int r = 0; r < 8; ++r)
      scores[(8 * half + r) * S_ + col] = c[r] * 0.125f + nm;
  }
  __syncthreads();

  // ---- Phase 2: dual softmax + adj/dist fusion (16 threads per row) ----
  const int row = t >> 4, part = t & 15;
  const float* drow = dist + ((size_t)b * S_ + q0 + row) * S_;
  const float* arow = adj + ((size_t)b * S_ + q0 + row) * S_;
  const float* mrow = mask + (size_t)b * S_;

  float mS = -3.4e38f, mD = -3.4e38f;
  for (int j = 0; j < S_ / 16; ++j) {
    int cc = part + j * 16;
    float s = scores[row * S_ + cc];
    float d = drow[cc] + mrow[cc] * -1e9f;
    mS = fmaxf(mS, s);
    mD = fmaxf(mD, d);
  }
  red[0][row][part] = mS; red[1][row][part] = mD;
  __syncthreads();
#pragma unroll
  for (int p = 0; p < 16; ++p) { mS = fmaxf(mS, red[0][row][p]); mD = fmaxf(mD, red[1][row][p]); }

  float sS = 0.f, sD = 0.f;
  for (int j = 0; j < S_ / 16; ++j) {
    int cc = part + j * 16;
    sS += __expf(scores[row * S_ + cc] - mS);
    sD += __expf(drow[cc] + mrow[cc] * -1e9f - mD);
  }
  __syncthreads();
  red[0][row][part] = sS; red[1][row][part] = sD;
  __syncthreads();
  sS = 0.f; sD = 0.f;
#pragma unroll
  for (int p = 0; p < 16; ++p) { sS += red[0][row][p]; sD += red[1][row][p]; }
  float invS = 1.f / sS, invD = 1.f / sD;

  float* aout = attn_out + ((size_t)bh * S_ + q0 + row) * S_;
  for (int j = 0; j < S_ / 16; ++j) {
    int cc = part + j * 16;
    float p1 = __expf(scores[row * S_ + cc] - mS) * invS;
    float p2 = __expf(drow[cc] + mrow[cc] * -1e9f - mD) * invD;
    float v = p1 * arow[cc] * p2;
    aout[cc] = v;
    pAttn[row * S_ + cc] = (bf16_t)v;
  }
  __syncthreads();

  // ---- Phase 3: ctx = attn @ vh (wave w: K slice [w*128, w*128+128)) ----
  v8f cacc[4] = {};
  const int kbase = wave * 128;
  for (int ks = 0; ks < 128; ks += 32) {
    v16bf a = frag_from_bf16(&pAttn[l16 * S_ + kbase + ks], half);
#pragma unroll
    for (int nt = 0; nt < 4; ++nt) {
      const bf16_t* vrow = vhT + ((size_t)bh * DP_ + nt * 16 + l16) * S_ + kbase + ks;
      v16bf bb = frag_from_bf16(vrow, half);
      cacc[nt] = wmma_bf16f32(a, bb, cacc[nt]);
    }
  }
  __syncthreads();                 // pAttn reads done; safe to alias ctxpart
#pragma unroll
  for (int nt = 0; nt < 4; ++nt)
#pragma unroll
    for (int r = 0; r < 8; ++r)
      ctxpart[wave * (16 * DP_) + (8 * half + r) * DP_ + nt * 16 + l16] = cacc[nt][r];
  __syncthreads();

  for (int i = t; i < 16 * DP_; i += 256) {
    float s = 0.f;
#pragma unroll
    for (int w2 = 0; w2 < 8; ++w2) s += ctxpart[w2 * (16 * DP_) + i];
    int r = i >> 6, dpc = i & 63;
    ctx[((size_t)b * S_ + q0 + r) * DM_ + h * DP_ + dpc] = (bf16_t)s;
  }
}

// ---------------------------------------------------------------------------
// Kernel 3: out = ctx(8192x512 bf16) @ Wo + bo, f32 output (d_out head).
// ---------------------------------------------------------------------------
__global__ void outproj_kernel(const bf16_t* __restrict__ ctx, const bf16_t* __restrict__ WoT,
                               const float* __restrict__ bo, float* __restrict__ out) {
  __shared__ bf16_t ldsB[2][64 * 32];
  const int n0 = blockIdx.x * 64;
  const int m0 = blockIdx.y * 128;
  const int t = threadIdx.x;
  const int wave = t >> 5, lane = t & 31;
  const int half = lane >> 4, l16 = lane & 15;

  const int rown = t >> 2, kc = (t & 3) * 8;
  const unsigned chunk = (unsigned)(rown * 64 + kc * 2);
  const unsigned ldsAddr0 = (unsigned)(uintptr_t)(&ldsB[0][0]) + chunk;
  const unsigned ldsAddr1 = (unsigned)(uintptr_t)(&ldsB[1][0]) + chunk;
  const bf16_t* wrow = WoT + (size_t)(n0 + rown) * DM_ + kc;

  async_copy_b128(ldsAddr0, wrow);
  int cur = 0;
  v8f acc[4] = {};
  for (int k0 = 0; k0 < DM_; k0 += 32, cur ^= 1) {
    if (k0 + 32 < DM_) {
      async_copy_b128(cur ? ldsAddr0 : ldsAddr1, wrow + k0 + 32);
      wait_async_le1();
    } else {
      wait_async_le0();
    }
    __syncthreads();

    const bf16_t* arow = ctx + (size_t)(m0 + wave * 16 + l16) * DM_ + k0;
    v16bf a = frag_from_bf16(arow, half);
#pragma unroll
    for (int nt = 0; nt < 4; ++nt) {
      v16bf b = frag_from_bf16(&ldsB[cur][(nt * 16 + l16) * 32], half);
      acc[nt] = wmma_bf16f32(a, b, acc[nt]);
    }
    __syncthreads();
  }

#pragma unroll
  for (int nt = 0; nt < 4; ++nt) {
    int n = n0 + nt * 16 + l16;
    float bia = bo[n];
#pragma unroll
    for (int r = 0; r < 8; ++r) {
      int m = m0 + wave * 16 + 8 * half + r;
      out[(size_t)m * DM_ + n] = acc[nt][r] + bia;
    }
  }
}

// ---------------------------------------------------------------------------
extern "C" void kernel_launch(void* const* d_in, const int* in_sizes, int n_in,
                              void* d_out, int out_size, void* d_ws, size_t ws_size,
                              hipStream_t stream) {
  (void)in_sizes; (void)n_in; (void)out_size; (void)ws_size;
  const float* q    = (const float*)d_in[0];
  const float* k    = (const float*)d_in[1];
  const float* v    = (const float*)d_in[2];
  const float* mask = (const float*)d_in[3];
  const float* adj  = (const float*)d_in[4];
  const float* dist = (const float*)d_in[5];
  const float* Wq   = (const float*)d_in[6];
  const float* bq   = (const float*)d_in[7];
  const float* Wk   = (const float*)d_in[8];
  const float* bk   = (const float*)d_in[9];
  const float* Wv   = (const float*)d_in[10];
  const float* bv   = (const float*)d_in[11];
  const float* Wo   = (const float*)d_in[12];
  const float* bo   = (const float*)d_in[13];

  float* out = (float*)d_out;
  float* attn_out = out + (size_t)B_ * S_ * DM_;

  const size_t HE = (size_t)B_ * H_ * S_ * DP_;   // 4,194,304 elements
  const size_t WE = (size_t)DM_ * DM_;            //   262,144 elements
  bf16_t* qh  = (bf16_t*)d_ws;                    // (B,H,S,DP)
  bf16_t* kh  = qh + HE;                          // (B,H,S,DP)
  bf16_t* vhT = kh + HE;                          // (B,H,DP,S)
  bf16_t* ctx = vhT + HE;                         // (B,S,DM)
  bf16_t* WqT = ctx + HE;                         // 4x (n,k) bf16 weights
  bf16_t* WkT = WqT + WE;
  bf16_t* WvT = WkT + WE;
  bf16_t* WoT = WvT + WE;
  bf16_t* qb  = WoT + WE;                         // (B,S,D) bf16 inputs
  bf16_t* kb  = qb + HE;
  bf16_t* vb  = kb + HE;
  // ws usage: ~58 MB

  convert_bf16<<<dim3((B_ * S_ * DM_) / (4 * 256), 1, 3), 256, 0, stream>>>(
      q, k, v, qb, kb, vb);
  prep_weights<<<dim3(DM_ / 64, DM_ / 64, 4), 256, 0, stream>>>(
      Wq, Wk, Wv, Wo, WqT, WkT, WvT, WoT);

  dim3 gProj(DM_ / 64, (B_ * S_) / 128);
  proj_kernel<<<gProj, 256, 0, stream>>>(qb, WqT, bq, qh, 0);
  proj_kernel<<<gProj, 256, 0, stream>>>(kb, WkT, bk, kh, 0);
  proj_kernel<<<gProj, 256, 0, stream>>>(vb, WvT, bv, vhT, 1);

  attn_kernel<<<dim3(S_ / 16, H_, B_), 256, 0, stream>>>(qh, kh, vhT, mask, adj, dist,
                                                         attn_out, ctx);

  outproj_kernel<<<gProj, 256, 0, stream>>>(ctx, WoT, bo, out);
}
